// PointPillarScatter_85229331022238
// MI455X (gfx1250) — matrix operand order repaired and checked by reading, not compiled, not run on hardware
//
#include <hip/hip_runtime.h>
#include <hip/hip_bf16.h>

#define NXg 432
#define NYg 496
#define Cg 64
#define KTOP 2800
#define Bg 4
#define Pg 12000
#define NCELLS (NXg*NYg)
#define YT 4                 // output rows per workgroup (4 waves)

typedef __attribute__((ext_vector_type(16))) __bf16 v16bf;
typedef __attribute__((ext_vector_type(8)))  float  v8f;
typedef __attribute__((ext_vector_type(4)))  float  f32x4;

static __device__ __forceinline__ unsigned short f2bf(float f) {
  unsigned u = __float_as_uint(f);
  unsigned r = ((u >> 16) & 1u) + 0x7FFFu;   // round-to-nearest-even
  return (unsigned short)((u + r) >> 16);
}

// ---------------------------------------------------------------------------
// 1) Scatter pillar features (f32) into channel-last bf16 BEV grid [B][cell][64]
// ---------------------------------------------------------------------------
__global__ void k_scatter(const float* __restrict__ pf, const int* __restrict__ vc,
                          unsigned short* __restrict__ grid, int* __restrict__ cells) {
  int i = blockIdx.x * blockDim.x + threadIdx.x;
  if (i >= Bg * Pg) return;
  int b = vc[i * 4 + 0];
  int y = vc[i * 4 + 2];
  int x = vc[i * 4 + 3];
  int cell = y * NXg + x;
  cells[i] = cell;
  const float* src = pf + (size_t)i * Cg;
  unsigned* dst = (unsigned*)(grid + ((size_t)b * NCELLS + cell) * Cg);
#pragma unroll
  for (int c = 0; c < Cg; c += 2) {
    unsigned lo = f2bf(src[c]);
    unsigned hi = f2bf(src[c + 1]);
    dst[c >> 1] = lo | (hi << 16);
  }
}

// ---------------------------------------------------------------------------
// 2) Pack conv1 weights (BN scale folded) into the 16-bit A-matrix (16x32)
//    VGPR layout (contiguous k, matched by B): lane L -> M=L%16; half h -> Klocal = h + (L>=16?16:0)
//    Global K order: k = tap*64 + c, tap = ky*3+kx
// ---------------------------------------------------------------------------
__global__ void k_packw(const float* __restrict__ w1, const float* __restrict__ g1,
                        const float* __restrict__ v1, unsigned short* __restrict__ wA) {
  int lane = threadIdx.x;  // 32 threads, 1 block
  int M = lane & 15;
  for (int ks = 0; ks < 18; ++ks) {
    for (int h = 0; h < 16; ++h) {
      int Kl = h + (lane >= 16 ? 16 : 0);
      int k = ks * 32 + Kl;
      int tap = k >> 6;
      int c = k & 63;
      int ky = tap / 3, kx = tap % 3;
      float val = 0.f;
      if (M < 8) {
        float sc = g1[M] * rsqrtf(v1[M] + 1e-3f);
        val = w1[((M * Cg + c) * 3 + ky) * 3 + kx] * sc;
      }
      wA[((size_t)ks * 32 + lane) * 16 + h] = f2bf(val);
    }
  }
}

// ---------------------------------------------------------------------------
// 3) conv1 as implicit-im2col GEMM on v_wmma_f32_16x16x32_bf16.
//    4 waves per workgroup; each wave owns one output row of a 16-wide x tile.
//    The 6-row halo (YT+2) is staged ONCE into LDS via the CDNA5 async
//    global->LDS path (GLOBAL_LOAD_ASYNC_TO_LDS_B128, ASYNCcnt), then each
//    wave runs K = 576 = 18 WMMA steps reading its 3-row window from LDS.
// ---------------------------------------------------------------------------
__global__ __launch_bounds__(128) void k_conv1(
    const unsigned short* __restrict__ grid, const unsigned short* __restrict__ wA,
    const float* __restrict__ b1, const float* __restrict__ m1,
    const float* __restrict__ g1, const float* __restrict__ v1,
    float* __restrict__ h1) {
  __shared__ alignas(16) unsigned short tile[(YT + 2) * 18 * Cg];  // 13824 B
  int bid = blockIdx.x;
  int xt = bid % 27;
  int yt = (bid / 27) % (NYg / YT);
  int b  = bid / (27 * (NYg / YT));
  int x0 = xt * 16;
  int y0 = yt * YT;
  int tid = threadIdx.x;

  // Stage (YT+2) rows x 18 cells x 64ch bf16 into LDS, zero-padded at borders.
  // In-bounds chunks use async global->LDS DMA (no VGPR round trip).
  unsigned lds_base = (unsigned)(uintptr_t)tile;  // low 32 bits = LDS offset
  for (int j = tid; j < (YT + 2) * 18 * 8; j += 128) {
    int r  = j / 144;
    int t  = j - r * 144;
    int cx = t >> 3;
    int ch = t & 7;
    int gy = y0 - 1 + r;
    int gx = x0 - 1 + cx;
    unsigned ldsa = lds_base + (unsigned)(j * 16);
    if (gy >= 0 && gy < NYg && gx >= 0 && gx < NXg) {
      const void* gp = (const void*)(grid +
          ((size_t)b * NCELLS + (size_t)gy * NXg + gx) * Cg + ch * 8);
      asm volatile("global_load_async_to_lds_b128 %0, %1, off"
                   :: "v"(ldsa), "v"(gp) : "memory");
    } else {
      f32x4 z = {0.f, 0.f, 0.f, 0.f};
      ((f32x4*)tile)[j] = z;
    }
  }
  asm volatile("s_wait_asynccnt 0x0" ::: "memory");
  __syncthreads();

  int wv   = tid >> 5;        // wave id 0..3 -> output row y0+wv
  int lane = tid & 31;
  int y    = y0 + wv;
  v8f acc = {};
  int nloc = lane & 15;
  int hi16 = (lane >> 4) << 4;  // +16 channel offset for lanes 16..31 (B layout)
#pragma unroll
  for (int ks = 0; ks < 18; ++ks) {
    int tap = ks >> 1;
    int cbase = (ks & 1) << 5;
    int ky = tap / 3, kx = tap % 3;
    union { f32x4 f4[2]; v16bf v; } A, Bm;
    const f32x4* ap = (const f32x4*)(wA + ((size_t)ks * 32 + lane) * 16);
    A.f4[0] = ap[0];
    A.f4[1] = ap[1];
    // B fragment: lane -> N = lane%16; 16 contiguous channels at cbase+hi16
    const f32x4* bp = (const f32x4*)(tile + ((wv + ky) * 18 + nloc + kx) * Cg + cbase + hi16);
    Bm.f4[0] = bp[0];
    Bm.f4[1] = bp[1];
    acc = __builtin_amdgcn_wmma_f32_16x16x32_bf16(false, A.v, false, Bm.v,
                                                  (short)0, acc, false, false);
  }

  // Epilogue: lanes 0..15 hold M=0..7 in acc[0..7]; bias + ReLU, store h1[b][m][y][x]
  if (lane < 16) {
#pragma unroll
    for (int r = 0; r < 8; ++r) {
      float sc = g1[r] * rsqrtf(v1[r] + 1e-3f);
      float bias = b1[r] - m1[r] * sc;
      float val = acc[r] + bias;
      h1[(((size_t)b * 8 + r) * NYg + y) * NXg + x0 + nloc] = val > 0.f ? val : 0.f;
    }
  }
}

// ---------------------------------------------------------------------------
// 4) conv2 (8 -> 1) + BN + ReLU -> heat map. L2-resident reads of h1.
// ---------------------------------------------------------------------------
__global__ void k_conv2(const float* __restrict__ h1, const float* __restrict__ w2,
                        const float* __restrict__ g2, const float* __restrict__ bb2,
                        const float* __restrict__ m2, const float* __restrict__ vv2,
                        float* __restrict__ heat) {
  int i = blockIdx.x * blockDim.x + threadIdx.x;
  if (i >= Bg * NCELLS) return;
  int b = i / NCELLS;
  int cell = i - b * NCELLS;
  int y = cell / NXg, x = cell - y * NXg;
  float sum = 0.f;
#pragma unroll
  for (int ky = 0; ky < 3; ++ky) {
    int yy = y + ky - 1;
    if (yy < 0 || yy >= NYg) continue;
#pragma unroll
    for (int kx = 0; kx < 3; ++kx) {
      int xx = x + kx - 1;
      if (xx < 0 || xx >= NXg) continue;
#pragma unroll
      for (int ic = 0; ic < 8; ++ic)
        sum += h1[(((size_t)b * 8 + ic) * NYg + yy) * NXg + xx] * w2[(ic * 3 + ky) * 3 + kx];
    }
  }
  float sc = g2[0] * rsqrtf(vv2[0] + 1e-3f);
  float val = sum * sc + (bb2[0] - m2[0] * sc);
  heat[i] = val > 0.f ? val : 0.f;
}

// ---------------------------------------------------------------------------
// 5) Top-K via 4-pass radix select (exact K-th key + tie quota)
// ---------------------------------------------------------------------------
__global__ void k_keys(const float* __restrict__ heat, const int* __restrict__ cells,
                       unsigned* __restrict__ keys) {
  int i = blockIdx.x * blockDim.x + threadIdx.x;
  if (i >= Bg * Pg) return;
  int b = i / Pg;
  float f = heat[(size_t)b * NCELLS + cells[i]];
  unsigned u = __float_as_uint(f);
  keys[i] = (u & 0x80000000u) ? ~u : (u | 0x80000000u);  // monotone map, larger = bigger
}

__global__ void k_init(unsigned* prefix, unsigned* need, unsigned* eqcnt) {
  int b = threadIdx.x;
  if (b < Bg) { prefix[b] = 0u; need[b] = KTOP; eqcnt[b] = 0u; }
}

__global__ void k_hist(const unsigned* __restrict__ keys, const unsigned* __restrict__ prefix,
                       unsigned* __restrict__ hist, int pass) {
  int i = blockIdx.x * blockDim.x + threadIdx.x;
  if (i >= Bg * Pg) return;
  int b = i / Pg;
  unsigned key = keys[i];
  bool match = (pass == 0) || ((key >> (32 - 8 * pass)) == prefix[b]);
  if (match) atomicAdd(&hist[b * 256 + ((key >> (24 - 8 * pass)) & 255u)], 1u);
}

__global__ void k_sel(const unsigned* __restrict__ hist, unsigned* prefix, unsigned* need) {
  int b = blockIdx.x;
  if (threadIdx.x != 0) return;
  unsigned nd = need[b], cum = 0;
  int d = 0;
  for (int j = 255; j >= 0; --j) {
    unsigned h = hist[b * 256 + j];
    if (cum + h >= nd) { d = j; break; }
    cum += h;
  }
  prefix[b] = (prefix[b] << 8) | (unsigned)d;
  need[b] = nd - cum;
}

// ---------------------------------------------------------------------------
// 6) Scatter selected pillars' original features into the zeroed output grid
// ---------------------------------------------------------------------------
__global__ void k_final(const unsigned* __restrict__ keys, const int* __restrict__ cells,
                        const unsigned* __restrict__ prefix, const unsigned* __restrict__ need,
                        unsigned* eqcnt, const float* __restrict__ pf,
                        float* __restrict__ out) {
  int i = blockIdx.x * blockDim.x + threadIdx.x;
  if (i >= Bg * Pg) return;
  int b = i / Pg;
  unsigned key = keys[i], T = prefix[b];
  bool take = key > T;
  if (key == T) take = (atomicAdd(&eqcnt[b], 1u) < need[b]);
  if (!take) return;
  int cell = cells[i];
  const float* src = pf + (size_t)i * Cg;
  float* dst = out + (size_t)b * Cg * NCELLS + cell;
#pragma unroll
  for (int c = 0; c < Cg; ++c) dst[(size_t)c * NCELLS] = src[c];
}

// ---------------------------------------------------------------------------
extern "C" void kernel_launch(void* const* d_in, const int* in_sizes, int n_in,
                              void* d_out, int out_size, void* d_ws, size_t ws_size,
                              hipStream_t stream) {
  (void)in_sizes; (void)n_in; (void)out_size; (void)ws_size;
  const float* pf = (const float*)d_in[0];
  const int*   vc = (const int*)d_in[1];
  const float* w1 = (const float*)d_in[2];
  const float* g1 = (const float*)d_in[3];
  const float* b1 = (const float*)d_in[4];
  const float* m1 = (const float*)d_in[5];
  const float* v1 = (const float*)d_in[6];
  const float* w2 = (const float*)d_in[7];
  const float* g2 = (const float*)d_in[8];
  const float* b2 = (const float*)d_in[9];
  const float* m2 = (const float*)d_in[10];
  const float* v2 = (const float*)d_in[11];
  float* out = (float*)d_out;

  char* ws = (char*)d_ws;
  size_t off = 0;
  auto alloc = [&](size_t bytes) {
    char* p = ws + off;
    off = (off + bytes + 255) & ~(size_t)255;
    return p;
  };
  unsigned short* grid = (unsigned short*)alloc((size_t)Bg * NCELLS * Cg * 2);
  unsigned short* wA   = (unsigned short*)alloc((size_t)18 * 32 * 16 * 2);
  float*    h1     = (float*)alloc((size_t)Bg * 8 * NCELLS * 4);
  float*    heat   = (float*)alloc((size_t)Bg * NCELLS * 4);
  int*      cells  = (int*)alloc((size_t)Bg * Pg * 4);
  unsigned* keys   = (unsigned*)alloc((size_t)Bg * Pg * 4);
  unsigned* hist   = (unsigned*)alloc((size_t)Bg * 256 * 4);
  unsigned* prefix = (unsigned*)alloc((size_t)Bg * 4);
  unsigned* need   = (unsigned*)alloc((size_t)Bg * 4);
  unsigned* eqcnt  = (unsigned*)alloc((size_t)Bg * 4);

  hipMemsetAsync(grid, 0, (size_t)Bg * NCELLS * Cg * 2, stream);
  hipMemsetAsync(out, 0, (size_t)Bg * Cg * NCELLS * 4, stream);

  k_scatter<<<(Bg * Pg + 255) / 256, 256, 0, stream>>>(pf, vc, grid, cells);
  k_packw<<<1, 32, 0, stream>>>(w1, g1, v1, wA);
  k_conv1<<<Bg * (NYg / YT) * 27, 128, 0, stream>>>(grid, wA, b1, m1, g1, v1, h1);
  k_conv2<<<(Bg * NCELLS + 255) / 256, 256, 0, stream>>>(h1, w2, g2, b2, m2, v2, heat);
  k_keys<<<(Bg * Pg + 255) / 256, 256, 0, stream>>>(heat, cells, keys);
  k_init<<<1, 32, 0, stream>>>(prefix, need, eqcnt);
  for (int pass = 0; pass < 4; ++pass) {
    hipMemsetAsync(hist, 0, (size_t)Bg * 256 * 4, stream);
    k_hist<<<(Bg * Pg + 255) / 256, 256, 0, stream>>>(keys, prefix, hist, pass);
    k_sel<<<Bg, 1, 0, stream>>>(hist, prefix, need);
  }
  k_final<<<(Bg * Pg + 255) / 256, 256, 0, stream>>>(keys, cells, prefix, need, eqcnt, pf, out);
}
